// CopyHead_86535001079875
// MI455X (gfx1250) — matrix operand order repaired and checked by reading, not compiled, not run on hardware
//
#include <hip/hip_runtime.h>

// ---------------- types for CDNA5 WMMA ----------------
typedef __attribute__((ext_vector_type(16))) __bf16 v16bf;
typedef __attribute__((ext_vector_type(8)))  float  v8f;

union FragBF {
    v16bf v;
    uint4 q[2];
};

#define NB 32
#define NS 512
#define ND 1024
#define NV 50257
#define NVS 50769            // NV + NS
#define LOG_TINY (-80.5904783f)   // logf(1e-35f)

static __device__ __forceinline__ unsigned short f2bf_sw(float f) {
    union { float f; unsigned int u; } c; c.f = f;
    unsigned int u = c.u;
    unsigned int r = (u + 0x7FFFu + ((u >> 16) & 1u)) >> 16;   // RNE truncate to bf16
    return (unsigned short)r;
}

// packed f32x2 -> bf16x2 : single v_cvt_pk_bf16_f32 when the builtin exists
static __device__ __forceinline__ unsigned int pack2(float lo, float hi) {
#if __has_builtin(__builtin_amdgcn_cvt_pk_bf16_f32)
    auto p = __builtin_amdgcn_cvt_pk_bf16_f32(lo, hi);
    unsigned int r;
    __builtin_memcpy(&r, &p, sizeof(r));
    return r;
#else
    return (unsigned int)f2bf_sw(lo) | ((unsigned int)f2bf_sw(hi) << 16);
#endif
}

static __device__ __forceinline__ float logaddexpf_(float a, float b) {
    float mx = fmaxf(a, b), mn = fminf(a, b);
    if (mx == -__builtin_inff()) return mx;
    return mx + log1pf(expf(mn - mx));
}

// =====================================================================
// Kernel A: gen_scores[b,v] = h[b,:] @ W_gen[:,v] + b_gen[v]
// Memory-bound on the 206MB W_gen stream. K-step 64: 2 WMMAs per barrier.
// Block: 256 thr (8 waves). wave -> (mtile = w&1, ntile = w>>1), 64 cols/blk.
// =====================================================================
__global__ __launch_bounds__(256)
void gen_scores_kernel(const float* __restrict__ h,
                       const float* __restrict__ Wg,
                       const float* __restrict__ bg,
                       float* __restrict__ gout /* [NB][NV] */) {
    __shared__ unsigned short Ah[32][72];   // 32 rows(batch) x 64 k, stride 72 (144B, 16B mult)
    __shared__ unsigned short Bt[64][72];   // 64 cols(v)    x 64 k, transposed W tile

    const int tid  = threadIdx.x;
    const int lane = tid & 31;
    const int wave = tid >> 5;
    const int mt   = wave & 1;       // 2 M tiles cover B=32 rows
    const int ntl  = wave >> 1;      // 4 N tiles of 16 cols
    const int v0   = blockIdx.x * 64;

    v8f acc = {0.f, 0.f, 0.f, 0.f, 0.f, 0.f, 0.f, 0.f};

    const int q  = tid & 15;         // k-group (4 k) for h staging
    const int m0 = tid >> 4;         // row for h staging
    const int g  = tid & 15;         // n-group (4 v) for W staging
    const int kk0 = tid >> 4;        // k row for W staging

    for (int kc = 0; kc < 16; ++kc) {
        const int k0 = kc * 64;
        // ---- stage h chunk: 32 x 64, float4 loads + packed uint2 stores ----
#pragma unroll
        for (int mi = 0; mi < 2; ++mi) {
            const int m = m0 + mi * 16;
            const float4 f = *reinterpret_cast<const float4*>(&h[m * ND + k0 + q * 4]);
            *reinterpret_cast<uint2*>(&Ah[m][q * 4]) =
                make_uint2(pack2(f.x, f.y), pack2(f.z, f.w));
        }
        // ---- stage W tile transposed: 64 k x 64 v (branchless tail guard) ----
#pragma unroll
        for (int ki = 0; ki < 4; ++ki) {
            const int kk = kk0 + ki * 16;
            const float* row = Wg + (size_t)(k0 + kk) * NV;
            if (kc < 15 && ki == 0)
                __builtin_prefetch(row + 64 * (size_t)NV + v0 + g * 4, 0, 1);
            float w[4];
#pragma unroll
            for (int i = 0; i < 4; ++i) {
                const int v  = v0 + g * 4 + i;
                const int vc = (v < NV) ? v : (NV - 1);
                w[i] = row[vc];
                if (v >= NV) w[i] = 0.f;
            }
            const unsigned int p01 = pack2(w[0], w[1]);
            const unsigned int p23 = pack2(w[2], w[3]);
            Bt[g * 4 + 0][kk] = (unsigned short)(p01 & 0xffffu);
            Bt[g * 4 + 1][kk] = (unsigned short)(p01 >> 16);
            Bt[g * 4 + 2][kk] = (unsigned short)(p23 & 0xffffu);
            Bt[g * 4 + 3][kk] = (unsigned short)(p23 >> 16);
        }
        __syncthreads();

        const int row = mt * 16 + (lane & 15);
        const int kb  = (lane >> 4) * 8;
        const int nn  = ntl * 16 + (lane & 15);
        const int kb2 = (lane >> 4) * 8;
        FragBF a0, a1, b0, b1;
        a0.q[0] = *reinterpret_cast<const uint4*>(&Ah[row][kb]);
        a0.q[1] = *reinterpret_cast<const uint4*>(&Ah[row][kb + 16]);
        a1.q[0] = *reinterpret_cast<const uint4*>(&Ah[row][32 + kb]);
        a1.q[1] = *reinterpret_cast<const uint4*>(&Ah[row][32 + kb + 16]);
        b0.q[0] = *reinterpret_cast<const uint4*>(&Bt[nn][kb2]);
        b0.q[1] = *reinterpret_cast<const uint4*>(&Bt[nn][kb2 + 16]);
        b1.q[0] = *reinterpret_cast<const uint4*>(&Bt[nn][32 + kb2]);
        b1.q[1] = *reinterpret_cast<const uint4*>(&Bt[nn][32 + kb2 + 16]);

        acc = __builtin_amdgcn_wmma_f32_16x16x32_bf16(false, a0.v, false, b0.v,
                                                      (short)0, acc, false, false);
        acc = __builtin_amdgcn_wmma_f32_16x16x32_bf16(false, a1.v, false, b1.v,
                                                      (short)0, acc, false, false);
        __syncthreads();
    }

    const int v = v0 + ntl * 16 + (lane & 15);
    if (v < NV) {
        const float bgv = bg[v];
#pragma unroll
        for (int r = 0; r < 8; ++r) {
            const int m = mt * 16 + (lane >> 4) * 8 + r;
            gout[(size_t)m * NV + v] = acc[r] + bgv;
        }
    }
}

// =====================================================================
// Kernel B: copy_scores[b,s] = sum_e tanh(enc[b,s,:]@W_copy[:,e]+bc[e])*h[b,e]
// Compute-bound (34.4 GFLOP) -> bf16 WMMA, fp32 accumulate, fused epilogue.
// Block: 256 thr (8 waves), 64 rows of one batch. wave -> (mtile=w&3, ntl=w>>2).
// enc block converted to bf16 in LDS once; W tiles (64k x 32e) per round.
// =====================================================================
__global__ __launch_bounds__(256)
void copy_scores_kernel(const float* __restrict__ enc,
                        const float* __restrict__ Wc,
                        const float* __restrict__ bc,
                        const float* __restrict__ h,
                        float* __restrict__ cs /* [NB][NS] */) {
    __shared__ unsigned short A[64][1032];   // 64 rows x 1024 k, stride 1032 (2064B, 16B mult)
    __shared__ unsigned short Bt[32][72];    // 32 cols(e) x 64 k transposed
    __shared__ float red[8][16];

    const int tid  = threadIdx.x;
    const int lane = tid & 31;
    const int wave = tid >> 5;
    const int mt   = wave & 3;
    const int ntl  = wave >> 2;
    const int b    = blockIdx.x >> 3;
    const int s0   = (blockIdx.x & 7) * 64;

    // ---- phase 0: enc block -> bf16 LDS (float4 loads, packed stores) ----
    const float* encb = enc + (size_t)(b * NS + s0) * ND;
    for (int e4 = tid; e4 < 64 * 256; e4 += 256) {
        const int r = e4 >> 8, k = (e4 & 255) * 4;
        const float4 f = *reinterpret_cast<const float4*>(&encb[(size_t)r * ND + k]);
        *reinterpret_cast<uint2*>(&A[r][k]) =
            make_uint2(pack2(f.x, f.y), pack2(f.z, f.w));
    }

    float rowsum[8];
#pragma unroll
    for (int r = 0; r < 8; ++r) rowsum[r] = 0.f;
    __syncthreads();

    const int g   = tid & 7;    // e-group of 4 for W staging
    const int kk0 = tid >> 3;   // k row (0..31) for W staging

    for (int nb = 0; nb < 32; ++nb) {
        const int e0 = nb * 32;
        v8f acc = {0.f, 0.f, 0.f, 0.f, 0.f, 0.f, 0.f, 0.f};
        for (int kc = 0; kc < 16; ++kc) {
            const int k0 = kc * 64;
            // stage W tile transposed: 64 k x 32 e
#pragma unroll
            for (int ki = 0; ki < 2; ++ki) {
                const int kk = kk0 + ki * 32;
                const float* wrow = Wc + (size_t)(k0 + kk) * ND + e0;
                if (kc < 15 && ki == 0)
                    __builtin_prefetch(wrow + 64 * ND + g * 4, 0, 1);
                const float4 w4 = *reinterpret_cast<const float4*>(&wrow[g * 4]);
                const unsigned int p01 = pack2(w4.x, w4.y);
                const unsigned int p23 = pack2(w4.z, w4.w);
                Bt[g * 4 + 0][kk] = (unsigned short)(p01 & 0xffffu);
                Bt[g * 4 + 1][kk] = (unsigned short)(p01 >> 16);
                Bt[g * 4 + 2][kk] = (unsigned short)(p23 & 0xffffu);
                Bt[g * 4 + 3][kk] = (unsigned short)(p23 >> 16);
            }
            __syncthreads();

            const int row = mt * 16 + (lane & 15);
            const int kb  = (lane >> 4) * 8;
            const int nn  = ntl * 16 + (lane & 15);
            const int kb2 = (lane >> 4) * 8;
            FragBF a0, a1, b0, b1;
            a0.q[0] = *reinterpret_cast<const uint4*>(&A[row][k0 + kb]);
            a0.q[1] = *reinterpret_cast<const uint4*>(&A[row][k0 + kb + 16]);
            a1.q[0] = *reinterpret_cast<const uint4*>(&A[row][k0 + 32 + kb]);
            a1.q[1] = *reinterpret_cast<const uint4*>(&A[row][k0 + 32 + kb + 16]);
            b0.q[0] = *reinterpret_cast<const uint4*>(&Bt[nn][kb2]);
            b0.q[1] = *reinterpret_cast<const uint4*>(&Bt[nn][kb2 + 16]);
            b1.q[0] = *reinterpret_cast<const uint4*>(&Bt[nn][32 + kb2]);
            b1.q[1] = *reinterpret_cast<const uint4*>(&Bt[nn][32 + kb2 + 16]);

            acc = __builtin_amdgcn_wmma_f32_16x16x32_bf16(false, a0.v, false, b0.v,
                                                          (short)0, acc, false, false);
            acc = __builtin_amdgcn_wmma_f32_16x16x32_bf16(false, a1.v, false, b1.v,
                                                          (short)0, acc, false, false);
            __syncthreads();
        }
        // fused epilogue: +bias, tanh, dot with h  (C layout: N=lane&15, M=(lane>>4)*8+r)
        const int nidx = e0 + ntl * 16 + (lane & 15);
        const float hv  = h[b * ND + nidx];
        const float bcv = bc[nidx];
#pragma unroll
        for (int r = 0; r < 8; ++r)
            rowsum[r] += tanhf(acc[r] + bcv) * hv;
    }

    // reduce partial dots across the 16 n-lanes of each half-wave
#pragma unroll
    for (int r = 0; r < 8; ++r)
        for (int msk = 1; msk <= 8; msk <<= 1)
            rowsum[r] += __shfl_xor(rowsum[r], msk, 32);
    if ((lane & 15) == 0) {
        const int half = lane >> 4;
#pragma unroll
        for (int r = 0; r < 8; ++r) red[wave][half * 8 + r] = rowsum[r];
    }
    __syncthreads();
    if (tid < 64) {
        const int mtl = tid >> 4, rl = tid & 15;
        cs[b * NS + s0 + mtl * 16 + rl] = red[mtl][rl] + red[mtl + 4][rl];
    }
}

// =====================================================================
// Kernel C: in-place log_softmax over concat([gen_scores, copy_scores+log(mask)])
// One block per batch row.
// =====================================================================
__global__ __launch_bounds__(512)
void logsoftmax_kernel(const unsigned char* __restrict__ mask,
                       const float* __restrict__ cs,
                       float* __restrict__ out) {
    __shared__ float sred[512];
    const int b = blockIdx.x, tid = threadIdx.x;
    float* gen = out + (size_t)NB * NVS + (size_t)b * NV;
    float* clp = out + (size_t)NB * NVS + (size_t)NB * NV + (size_t)b * NS;

    const float xs = cs[b * NS + tid] +
                     logf((mask[b * NS + tid] ? 1.0f : 0.0f) + 1e-35f);

    float mx = -__builtin_inff();
    for (int v = tid; v < NV; v += 512) mx = fmaxf(mx, gen[v]);
    mx = fmaxf(mx, xs);
    sred[tid] = mx; __syncthreads();
    for (int off = 256; off > 0; off >>= 1) {
        if (tid < off) sred[tid] = fmaxf(sred[tid], sred[tid + off]);
        __syncthreads();
    }
    mx = sred[0];
    __syncthreads();

    float sum = 0.f;
    for (int v = tid; v < NV; v += 512) sum += expf(gen[v] - mx);
    sum += expf(xs - mx);
    sred[tid] = sum; __syncthreads();
    for (int off = 256; off > 0; off >>= 1) {
        if (tid < off) sred[tid] += sred[tid + off];
        __syncthreads();
    }
    const float ls = logf(sred[0]) + mx;

    for (int v = tid; v < NV; v += 512) gen[v] = gen[v] - ls;
    clp[tid] = xs - ls;
}

// =====================================================================
// Kernel D: associative reformulation of the S-step scan.
// Token/id/log-prob rows staged via ASYNC global->LDS copies, overlapped
// with the 50257-element row copy, then s_wait_asynccnt + barrier.
// =====================================================================
__global__ __launch_bounds__(512)
void scan_kernel(const int* __restrict__ tok,
                 const int* __restrict__ s2t,
                 float* __restrict__ out) {
    __shared__ int   tokS[512];
    __shared__ int   idsS[512];
    __shared__ float clpS[512];
    const int b = blockIdx.x, tid = threadIdx.x;
    const float* gen = out + (size_t)NB * NVS + (size_t)b * NV;
    const float* clp = out + (size_t)NB * NVS + (size_t)NB * NV + (size_t)b * NS;
    float* fl = out + (size_t)b * NVS;

    // async global -> LDS copies (ASYNCcnt), one dword per lane
    {
        unsigned lt = (unsigned)(unsigned long long)&tokS[tid];
        unsigned li = (unsigned)(unsigned long long)&idsS[tid];
        unsigned lc = (unsigned)(unsigned long long)&clpS[tid];
        const int*   gt = tok + b * NS + tid;
        const int*   gi = s2t + b * NS + tid;
        const float* gc = clp + tid;
        asm volatile("global_load_async_to_lds_b32 %0, %1, off"
                     :: "v"(lt), "v"(gt) : "memory");
        asm volatile("global_load_async_to_lds_b32 %0, %1, off"
                     :: "v"(li), "v"(gi) : "memory");
        asm volatile("global_load_async_to_lds_b32 %0, %1, off"
                     :: "v"(lc), "v"(gc) : "memory");
    }

    // overlap: copy generation part while async copies are in flight
    for (int v = tid; v < NV; v += 512) fl[v] = gen[v];

    asm volatile("s_wait_asynccnt 0x0" ::: "memory");
    __syncthreads();

    const int i  = tid;
    const int id = idsS[i];

    // scatter-combine into generation part (id 0..3 are PAD/UNK/START/END: no-op)
    if (id >= 4) {
        bool first = true;
        for (int j = 0; j < i; ++j)
            if (idsS[j] == id) { first = false; break; }
        if (first) {
            float m = clpS[i];
            for (int j = i + 1; j < NS; ++j)
                if (idsS[j] == id) m = fmaxf(m, clpS[j]);
            float s = 0.f;
            for (int j = i; j < NS; ++j)
                if (idsS[j] == id) s += expf(clpS[j] - m);
            const float lse = m + logf(s);
            fl[id] = logaddexpf_(fl[id], lse);
        }
    }

    // leftover part
    const int tk = tokS[i];
    float m2 = -__builtin_inff();
    for (int j = 0; j <= i; ++j)
        if (tokS[j] == tk) m2 = fmaxf(m2, clpS[j]);
    float s2 = 0.f;
    for (int j = 0; j <= i; ++j)
        if (tokS[j] == tk) s2 += expf(clpS[j] - m2);
    float val = m2 + logf(s2);

    bool fut = false;
    for (int j = i + 1; j < NS; ++j)
        if (tokS[j] == tk) { fut = true; break; }
    if (fut)     val += LOG_TINY;   // future duplicate -> masked out
    if (id >= 4) val += LOG_TINY;   // copy-able id -> leftover masked out
    fl[NV + i] = val;
}

// =====================================================================
extern "C" void kernel_launch(void* const* d_in, const int* in_sizes, int n_in,
                              void* d_out, int out_size, void* d_ws, size_t ws_size,
                              hipStream_t stream) {
    const int*           tokv = (const int*)d_in[0];
    const int*           s2t  = (const int*)d_in[1];
    const unsigned char* msk  = (const unsigned char*)d_in[2];
    const float*         enc  = (const float*)d_in[3];
    const float*         h    = (const float*)d_in[4];
    const float*         Wg   = (const float*)d_in[5];
    const float*         bg   = (const float*)d_in[6];
    const float*         Wc   = (const float*)d_in[7];
    const float*         bc   = (const float*)d_in[8];
    float* out = (float*)d_out;
    float* cs  = (float*)d_ws;   // NB*NS floats of scratch (64 KB)

    // raw gen scores straight into the generation_log_probs output region
    gen_scores_kernel<<<(NV + 63) / 64, 256, 0, stream>>>(h, Wg, bg,
                                                          out + (size_t)NB * NVS);
    copy_scores_kernel<<<NB * (NS / 64), 256, 0, stream>>>(enc, Wc, bc, h, cs);
    logsoftmax_kernel<<<NB, 512, 0, stream>>>(msk, cs, out);
    scan_kernel<<<NB, 512, 0, stream>>>(tokv, s2t, out);
}
